// HierAutoRegModel_64390149701897
// MI455X (gfx1250) — matrix operand order repaired and checked by reading, not compiled, not run on hardware
//
#include <hip/hip_runtime.h>
#include <hip/hip_bf16.h>

typedef __bf16 bf16;
typedef __attribute__((ext_vector_type(8)))  __bf16 v8bf;
typedef __attribute__((ext_vector_type(16))) __bf16 v16bf;
typedef __attribute__((ext_vector_type(8)))  float  v8f;

#define NN   2048   // batch
#define EE   512    // embedding
#define E3   1536   // 3*E
#define VV   1024   // vocab+1
#define UTT  64
#define NMT  6
#define MPT  10

__device__ __forceinline__ float sigf(float x) { return 1.0f / (1.0f + __expf(-x)); }

// ---------------------------------------------------------------------------
// Weight conversion: W[J,K] f32 row-major  ->  bf16 pre-swizzled WMMA B-operand
// layout for v_wmma_f32_16x16x32_bf16.  Tile = 32(K) x 16(J) = 512 bf16.
// Within a tile: lane = ((k>>4)&1)*16 + (j&15), element = k&15 (16 contiguous
// bf16 per lane -> one 32B coalesced read per lane at GEMM time).
// ---------------------------------------------------------------------------
__global__ void k_convert_swizzle(const float* __restrict__ W, bf16* __restrict__ out,
                                  int J, int K)
{
    int total = J * K;
    for (int idx = blockIdx.x * blockDim.x + threadIdx.x; idx < total;
         idx += gridDim.x * blockDim.x) {
        int j = idx / K;
        int k = idx - j * K;
        int kt = k >> 5;
        int jt = j >> 4;
        int lane = (((k >> 4) & 1) << 4) | (j & 15);
        int elem = k & 15;
        size_t o = (((size_t)kt * (J >> 4) + jt) * 32 + lane) * 16 + elem;
        out[o] = (bf16)W[idx];
    }
}

// ---------------------------------------------------------------------------
// Init: upper = sum of meaning embeddings, lower = 0, stop = 1,
// tok = one-hot(start token V-1) in bf16.
// ---------------------------------------------------------------------------
__global__ void k_init(const int* __restrict__ meanings, const float* __restrict__ emb,
                       float* __restrict__ upper, bf16* __restrict__ upper_b,
                       float* __restrict__ lower, bf16* __restrict__ lower_b,
                       float* __restrict__ stopv, bf16* __restrict__ tok)
{
    int total = NN * VV;
    for (int idx = blockIdx.x * blockDim.x + threadIdx.x; idx < total;
         idx += gridDim.x * blockDim.x) {
        int n = idx / VV;
        int c = idx - n * VV;
        tok[idx] = (bf16)((c == VV - 1) ? 1.0f : 0.0f);
        if (c < EE) {
            float s = 0.0f;
            #pragma unroll
            for (int t = 0; t < NMT; ++t)
                s += emb[(size_t)(meanings[n * NMT + t] + t * MPT) * EE + c];
            int ui = n * EE + c;
            upper[ui] = s;  upper_b[ui] = (bf16)s;
            lower[ui] = 0.0f; lower_b[ui] = (bf16)0.0f;
        }
        if (c == 0) stopv[n] = 1.0f;
    }
}

// ---------------------------------------------------------------------------
// Async global->LDS staging (CDNA5 GLOBAL_LOAD_ASYNC_TO_LDS_B128, ASYNCcnt),
// with a synchronous VGPR-bounce fallback if the builtin is unavailable.
// Builtin signature (per clang diagnostic): (int4 AS1*, int4 AS3*, Ii, Ii).
// generic->AS1 is bit-identical; generic->AS3 is the low 32 bits (LDS offset),
// so integer round-trips implement the addrspacecasts portably.
// ---------------------------------------------------------------------------
typedef __attribute__((__vector_size__(16))) int i128v;

#if defined(__gfx1250__) && __has_builtin(__builtin_amdgcn_global_load_async_to_lds_b128)
#define GEMM_ASYNC_COPY 1
#define CP16(d, s)                                                                      \
    __builtin_amdgcn_global_load_async_to_lds_b128(                                     \
        (__attribute__((address_space(1))) i128v*)(unsigned long long)(s),              \
        (__attribute__((address_space(3))) i128v*)(unsigned int)(unsigned long long)(d),\
        0, 0)
#if __has_builtin(__builtin_amdgcn_s_wait_asynccnt)
#define WAIT_ASYNC() __builtin_amdgcn_s_wait_asynccnt(0)
#else
#define WAIT_ASYNC() asm volatile("s_wait_asynccnt 0" ::: "memory")
#endif
#else
#define CP16(d, s) (*(uint4*)(d) = *(const uint4*)(s))
#define WAIT_ASYNC() ((void)0)
#endif

// ---------------------------------------------------------------------------
// WMMA GEMM:  C[M,J] = A[M,K](bf16) @ Bswz (pre-swizzled Wᵀ, bf16) + bias[J]
// Block: 256 threads = 8 waves.  Block tile 128Mx64N, K step 32.
// Each wave: one 16-row strip, 4 accumulators (16x64).
// Software pipeline: double-buffered LDS A-tile staged via async copy while
// the current chunk computes; all 4 B fragments gathered into registers
// before the 4 back-to-back WMMAs.  One barrier per K-chunk.
// ---------------------------------------------------------------------------
__global__ __launch_bounds__(256)
void k_gemm_bf16(const bf16* __restrict__ A, const bf16* __restrict__ Bswz,
                 const float* __restrict__ bias,
                 float* __restrict__ Cf, bf16* __restrict__ Cb,
                 int M, int J, int K)
{
    __shared__ bf16 sA[2][128][40];   // padded rows: conflict-free b128 LDS reads

    const int tid  = threadIdx.x;
    const int wave = tid >> 5;
    const int lane = tid & 31;
    const int m0   = blockIdx.y * 128;
    const int j0   = blockIdx.x * 64;
    const int Jt   = J >> 4;
    const int jt16 = j0 >> 4;

    // cooperative stage: 2 threads/row, 32B (16 bf16) per thread
    const int crow  = tid >> 1;
    const int chalf = tid & 1;
    const bf16* gsrc0 = A + (size_t)(m0 + crow) * K + chalf * 16;
    bf16* ldst0 = &sA[0][crow][chalf * 16];
    bf16* ldst1 = &sA[1][crow][chalf * 16];

    v8f acc[4] = {};

    const int arow  = wave * 16 + (lane & 15);
    const int ahalf = lane >> 4;
    const int KT    = K >> 5;

    // prologue: stage chunk 0 into buffer 0
    CP16(ldst0,     gsrc0);
    CP16(ldst0 + 8, gsrc0 + 8);
    WAIT_ASYNC();
    __syncthreads();

    for (int kt = 0; kt < KT; ++kt) {
        const int buf = kt & 1;
        // stage next chunk into the other buffer (async, overlapped with math)
        if (kt + 1 < KT) {
            const bf16* gs = gsrc0 + (kt + 1) * 32;
            bf16* ld = (buf ? ldst0 : ldst1);
            CP16(ld,     gs);
            CP16(ld + 8, gs + 8);
        }
        if (kt + 2 < KT)   // prefetch chunk kt+2 into L2 (global_prefetch_b8)
            __builtin_prefetch(gsrc0 + (kt + 2) * 32, 0, 1);

        // A fragment (matches B swizzle): lanes 0-15 K0-15 / lanes 16-31 K16-31
        // in elems 0-15.
        v8bf a0 = *(const v8bf*)(&sA[buf][arow][ahalf * 16]);
        v8bf a1 = *(const v8bf*)(&sA[buf][arow][ahalf * 16 + 8]);
        v16bf a = __builtin_shufflevector(a0, a1, 0,1,2,3,4,5,6,7,8,9,10,11,12,13,14,15);

        // gather all B fragments first, then issue WMMAs back-to-back
        v16bf bfrag[4];
        #pragma unroll
        for (int jt = 0; jt < 4; ++jt) {
            const bf16* bp = Bswz + (((size_t)kt * Jt + (jt16 + jt)) * 32 + lane) * 16;
            v8bf b0 = *(const v8bf*)(bp);
            v8bf b1 = *(const v8bf*)(bp + 8);
            bfrag[jt] = __builtin_shufflevector(b0, b1, 0,1,2,3,4,5,6,7,8,9,10,11,12,13,14,15);
        }
        #pragma unroll
        for (int jt = 0; jt < 4; ++jt)
            acc[jt] = __builtin_amdgcn_wmma_f32_16x16x32_bf16(
                false, a, false, bfrag[jt], (short)0, acc[jt], false, false);

        WAIT_ASYNC();      // next buffer fully in LDS for this wave
        __syncthreads();   // ... and for all waves
    }

    // Epilogue.  C/D layout: VGPR r -> row = (lane>>4)*8 + r, col = lane&15.
    const int erow = (lane >> 4) * 8;
    const int ecol = lane & 15;
    #pragma unroll
    for (int jt = 0; jt < 4; ++jt) {
        const int jc = j0 + jt * 16 + ecol;
        const float bv = bias[jc];
        #pragma unroll
        for (int r = 0; r < 8; ++r) {
            const int row = m0 + wave * 16 + erow + r;
            const float v = acc[jt][r] + bv;
            if (Cf) Cf[(size_t)row * J + jc] = v;
            if (Cb) Cb[(size_t)row * J + jc] = (bf16)v;
        }
    }
}

// ---------------------------------------------------------------------------
// Upper GRU update + stop gating.  gh already includes bhh_u (GEMM bias).
// gi_u is the constant row bih_u (upper input is always zero).
// ---------------------------------------------------------------------------
__global__ void k_upper_update(const float* __restrict__ gh, const float* __restrict__ bih_u,
                               float* __restrict__ upper, bf16* __restrict__ upper_b,
                               float* __restrict__ lower, bf16* __restrict__ lower_b,
                               const float* __restrict__ stopv)
{
    int idx = blockIdx.x * blockDim.x + threadIdx.x;
    if (idx >= NN * EE) return;
    int n = idx / EE;
    int e = idx - n * EE;
    size_t g = (size_t)n * E3;
    float r  = sigf(bih_u[e]          + gh[g + e]);
    float z  = sigf(bih_u[EE + e]     + gh[g + EE + e]);
    float nv = tanhf(bih_u[2 * EE + e] + r * gh[g + 2 * EE + e]);
    float u  = upper[idx];
    float nu = (1.0f - z) * nv + z * u;
    float s  = stopv[n];
    float u2 = s * nu + (1.0f - s) * u;
    float l2 = s * u2 + (1.0f - s) * lower[idx];
    upper[idx] = u2;  upper_b[idx] = (bf16)u2;
    lower[idx] = l2;  lower_b[idx] = (bf16)l2;
}

// ---------------------------------------------------------------------------
// Lower GRU update.  gi/gh already include their biases.
// ---------------------------------------------------------------------------
__global__ void k_lower_update(const float* __restrict__ gi, const float* __restrict__ gh,
                               float* __restrict__ lower, bf16* __restrict__ lower_b)
{
    int idx = blockIdx.x * blockDim.x + threadIdx.x;
    if (idx >= NN * EE) return;
    int n = idx / EE;
    int e = idx - n * EE;
    size_t g = (size_t)n * E3;
    float r  = sigf(gi[g + e]          + gh[g + e]);
    float z  = sigf(gi[g + EE + e]     + gh[g + EE + e]);
    float nv = tanhf(gi[g + 2 * EE + e] + r * gh[g + 2 * EE + e]);
    float l  = lower[idx];
    float l2 = (1.0f - z) * nv + z * l;
    lower[idx] = l2;  lower_b[idx] = (bf16)l2;
}

// ---------------------------------------------------------------------------
// stop[n] = sigmoid(dot(lower[n,:], Wstop) + bstop).  One wave per row.
// ---------------------------------------------------------------------------
__global__ __launch_bounds__(256)
void k_stop(const float* __restrict__ lower, const float* __restrict__ Wstop,
            const float* __restrict__ bstop, float* __restrict__ stopv)
{
    int wave = threadIdx.x >> 5;
    int lane = threadIdx.x & 31;
    int n = blockIdx.x * 8 + wave;
    const float* lr = lower + (size_t)n * EE;
    float s = 0.0f;
    for (int i = lane; i < EE; i += 32) s += lr[i] * Wstop[i];
    #pragma unroll
    for (int o = 16; o > 0; o >>= 1) s += __shfl_xor(s, o, 32);
    if (lane == 0) stopv[n] = sigf(s + bstop[0]);
}

// ---------------------------------------------------------------------------
// Softmax over V=1024 logits -> bf16 token feedback.  One block per row.
// ---------------------------------------------------------------------------
__global__ __launch_bounds__(256)
void k_softmax(const float* __restrict__ logits, bf16* __restrict__ tok)
{
    __shared__ float red[8];
    __shared__ float rowmax, rowsum;
    const int n = blockIdx.x;
    const int tid = threadIdx.x;
    const float* row = logits + (size_t)n * VV;

    float m = -3.0e38f;
    for (int i = tid; i < VV; i += 256) m = fmaxf(m, row[i]);
    #pragma unroll
    for (int o = 16; o > 0; o >>= 1) m = fmaxf(m, __shfl_xor(m, o, 32));
    if ((tid & 31) == 0) red[tid >> 5] = m;
    __syncthreads();
    if (tid == 0) {
        float v = red[0];
        for (int i = 1; i < 8; ++i) v = fmaxf(v, red[i]);
        rowmax = v;
    }
    __syncthreads();
    m = rowmax;

    float s = 0.0f;
    for (int i = tid; i < VV; i += 256) s += __expf(row[i] - m);
    #pragma unroll
    for (int o = 16; o > 0; o >>= 1) s += __shfl_xor(s, o, 32);
    if ((tid & 31) == 0) red[tid >> 5] = s;
    __syncthreads();
    if (tid == 0) {
        float v = 0.0f;
        for (int i = 0; i < 8; ++i) v += red[i];
        rowsum = v;
    }
    __syncthreads();
    const float inv = 1.0f / rowsum;
    for (int i = tid; i < VV; i += 256)
        tok[(size_t)n * VV + i] = (bf16)(__expf(row[i] - m) * inv);
}

// ---------------------------------------------------------------------------
extern "C" void kernel_launch(void* const* d_in, const int* in_sizes, int n_in,
                              void* d_out, int out_size, void* d_ws, size_t ws_size,
                              hipStream_t stream)
{
    (void)in_sizes; (void)n_in; (void)out_size; (void)ws_size;

    const int*   meanings   = (const int*)d_in[0];
    const float* embeddings = (const float*)d_in[1];
    /* d_in[2] = Wih_u : unused (upper RNN input is always zero) */
    const float* Whh_u = (const float*)d_in[3];
    const float* bih_u = (const float*)d_in[4];
    const float* bhh_u = (const float*)d_in[5];
    const float* Wih_l = (const float*)d_in[6];
    const float* Whh_l = (const float*)d_in[7];
    const float* bih_l = (const float*)d_in[8];
    const float* bhh_l = (const float*)d_in[9];
    const float* Wv2e  = (const float*)d_in[10];
    const float* bv2e  = (const float*)d_in[11];
    const float* We2v  = (const float*)d_in[12];
    const float* be2v  = (const float*)d_in[13];
    const float* Wstop = (const float*)d_in[14];
    const float* bstop = (const float*)d_in[15];
    float* out = (float*)d_out;

    // ---- workspace carve-up (256B aligned) ----
    char* ws = (char*)d_ws;
    auto carve = [&](size_t bytes) -> char* {
        char* p = ws;
        ws += (bytes + 255) & ~(size_t)255;
        return p;
    };
    bf16*  WhhuS  = (bf16*)carve((size_t)E3 * EE * 2);
    bf16*  WihlS  = (bf16*)carve((size_t)E3 * EE * 2);
    bf16*  WhhlS  = (bf16*)carve((size_t)E3 * EE * 2);
    bf16*  Wv2eS  = (bf16*)carve((size_t)EE * VV * 2);
    bf16*  We2vS  = (bf16*)carve((size_t)VV * EE * 2);
    float* upper  = (float*)carve((size_t)NN * EE * 4);
    float* lower  = (float*)carve((size_t)NN * EE * 4);
    bf16*  upper_b = (bf16*)carve((size_t)NN * EE * 2);
    bf16*  lower_b = (bf16*)carve((size_t)NN * EE * 2);
    bf16*  x_b     = (bf16*)carve((size_t)NN * EE * 2);
    bf16*  tok_b   = (bf16*)carve((size_t)NN * VV * 2);
    float* gh_u    = (float*)carve((size_t)NN * E3 * 4);
    float* gi_l    = (float*)carve((size_t)NN * E3 * 4);
    float* gh_l    = (float*)carve((size_t)NN * E3 * 4);
    float* stopv   = (float*)carve((size_t)NN * 4);

    // ---- one-time (per launch) weight conversion + swizzle ----
    auto conv = [&](const float* W, bf16* o, int J, int K) {
        int total = J * K;
        k_convert_swizzle<<<dim3((total + 255) / 256), dim3(256), 0, stream>>>(W, o, J, K);
    };
    conv(Whh_u, WhhuS, E3, EE);
    conv(Wih_l, WihlS, E3, EE);
    conv(Whh_l, WhhlS, E3, EE);
    conv(Wv2e,  Wv2eS, EE, VV);
    conv(We2v,  We2vS, VV, EE);

    k_init<<<dim3((NN * VV + 255) / 256), dim3(256), 0, stream>>>(
        meanings, embeddings, upper, upper_b, lower, lower_b, stopv, tok_b);

    auto gemm = [&](const bf16* A, const bf16* B, const float* bias,
                    float* Cf, bf16* Cb, int M, int J, int K) {
        k_gemm_bf16<<<dim3(J / 64, M / 128), dim3(256), 0, stream>>>(A, B, bias, Cf, Cb, M, J, K);
    };

    const dim3 eblk(256), egrd((NN * EE + 255) / 256);

    for (int t = 0; t < UTT; ++t) {
        // upper GRU
        gemm(upper_b, WhhuS, bhh_u, gh_u, nullptr, NN, E3, EE);
        k_upper_update<<<egrd, eblk, 0, stream>>>(gh_u, bih_u, upper, upper_b,
                                                  lower, lower_b, stopv);
        // token embedding: x = tok @ Wv2eᵀ + bv2e  (only needed as bf16 A-operand)
        gemm(tok_b, Wv2eS, bv2e, nullptr, x_b, NN, EE, VV);
        // lower GRU gate pre-activations
        gemm(x_b,     WihlS, bih_l, gi_l, nullptr, NN, E3, EE);
        gemm(lower_b, WhhlS, bhh_l, gh_l, nullptr, NN, E3, EE);
        k_lower_update<<<egrd, eblk, 0, stream>>>(gi_l, gh_l, lower, lower_b);
        // stop gate
        k_stop<<<dim3(NN / 8), dim3(256), 0, stream>>>(lower, Wstop, bstop, stopv);
        // logits -> d_out[t], softmax feedback -> tok
        float* logits = out + (size_t)t * NN * VV;
        gemm(lower_b, We2vS, be2v, logits, nullptr, NN, VV, EE);
        k_softmax<<<dim3(NN), dim3(256), 0, stream>>>(logits, tok_b);
    }
}
